// LoraAttention_48868137894052
// MI455X (gfx1250) — matrix operand order repaired and checked by reading, not compiled, not run on hardware
//
#include <hip/hip_runtime.h>
#include <hip/hip_bf16.h>

// ---------------------------------------------------------------------------
// LoRA attention forward for MI455X (gfx1250), wave32 + WMMA bf16 path,
// async global->LDS staging (ASYNCcnt) for GEMM/attention tiles.
// ---------------------------------------------------------------------------

typedef __attribute__((ext_vector_type(16))) __bf16 v16bf;
typedef __attribute__((ext_vector_type(8)))  float  v8f;

#define DIMC      1024
#define HEADS     16
#define HD        64
#define SEQ       2048
#define BATCH     2
#define NTOK      (BATCH * SEQ)        // 4096 tokens
#define ATT_SCALE 0.125f               // 64^-0.5
#define LORA_SC   0.1f                 // 1/R, R=10
#define RANK      10
#define NEG_BIG   (-1e30f)

union FragA { unsigned int u[8]; v16bf v; };

__device__ __forceinline__ v8f wmma_bf16(v16bf a, v16bf b, v8f c) {
  return __builtin_amdgcn_wmma_f32_16x16x32_bf16(false, a, false, b,
                                                 (short)0, c, false, false);
}

__device__ __forceinline__ unsigned short f2bf_raw(float f) {
  __hip_bfloat16 h = __float2bfloat16(f);
  return *reinterpret_cast<unsigned short*>(&h);
}

// Async 16B global -> LDS copy (per-lane), tracked by ASYNCcnt.
// LDS dest = low 32 bits of the flat pointer (wave-relative LDS offset).
__device__ __forceinline__ void async_ld_b128(void* lds_dst, const void* gsrc) {
  unsigned off = (unsigned)(size_t)lds_dst;
  asm volatile("global_load_async_to_lds_b128 %0, %1, off"
               :: "v"(off), "v"(gsrc) : "memory");
}
__device__ __forceinline__ void wait_async0() {
  asm volatile("s_wait_asynccnt 0x0" ::: "memory");
}

// ---------------------------------------------------------------------------
// 16-lane (half-wave) reductions via v_permlane16_b32 (pure VALU, no LDS).
// ---------------------------------------------------------------------------
#if __has_builtin(__builtin_amdgcn_permlane16)
__device__ __forceinline__ float pl16(float x, int s1, int s2) {
  int i = __float_as_int(x);
  i = __builtin_amdgcn_permlane16(i, i, s1, s2, false, false);
  return __int_as_float(i);
}
__device__ __forceinline__ float rmax16(float v) {
  v = fmaxf(v, pl16(v, 0x67452301, 0xEFCDAB89));  // xor 1
  v = fmaxf(v, pl16(v, 0x54761032, 0xDCFE98BA));  // xor 2
  v = fmaxf(v, pl16(v, 0x32107654, 0xBA98FEDC));  // xor 4
  v = fmaxf(v, pl16(v, 0xFEDCBA98, 0x76543210));  // xor 8
  return v;
}
__device__ __forceinline__ float rsum16(float v) {
  v += pl16(v, 0x67452301, 0xEFCDAB89);
  v += pl16(v, 0x54761032, 0xDCFE98BA);
  v += pl16(v, 0x32107654, 0xBA98FEDC);
  v += pl16(v, 0xFEDCBA98, 0x76543210);
  return v;
}
#else
__device__ __forceinline__ float rmax16(float v) {
  #pragma unroll
  for (int m = 1; m < 16; m <<= 1) v = fmaxf(v, __shfl_xor(v, m, 32));
  return v;
}
__device__ __forceinline__ float rsum16(float v) {
  #pragma unroll
  for (int m = 1; m < 16; m <<= 1) v += __shfl_xor(v, m, 32);
  return v;
}
#endif

// ---------------------------------------------------------------------------
// Prep: fp32 -> bf16 conversion
// ---------------------------------------------------------------------------
__global__ void cvt_bf16_kernel(const float* __restrict__ src,
                                __hip_bfloat16* __restrict__ dst, int n) {
  int i = blockIdx.x * 256 + threadIdx.x;
  if (i < n) dst[i] = __float2bfloat16(src[i]);
}

// ---------------------------------------------------------------------------
// Prep: build effective bf16 weights (LoRA rank-10 update folded in).
// ---------------------------------------------------------------------------
__global__ void build_weights_kernel(
    const float* __restrict__ w_qkv,
    const float* __restrict__ wq_base, const float* __restrict__ wqA,
    const float* __restrict__ wqB,
    const float* __restrict__ wv_base, const float* __restrict__ wvA,
    const float* __restrict__ wvB,
    const float* __restrict__ w_out,
    __hip_bfloat16* __restrict__ wqe, __hip_bfloat16* __restrict__ wke,
    __hip_bfloat16* __restrict__ wve, __hip_bfloat16* __restrict__ wob) {
  int idx = blockIdx.x * 256 + threadIdx.x;      // 0 .. 4M-1
  int seg = idx >> 20;
  int e   = idx & ((1 << 20) - 1);               // n*1024 + k
  int n   = e >> 10;
  int k   = e & 1023;
  if (seg == 0) {
    float acc = 0.f;
    #pragma unroll
    for (int r = 0; r < RANK; ++r) acc += wqB[n * RANK + r] * wqA[r * DIMC + k];
    wqe[e] = __float2bfloat16(w_qkv[e] + wq_base[e] + LORA_SC * acc);
  } else if (seg == 1) {
    wke[e] = __float2bfloat16(w_qkv[(1 << 20) + e]);
  } else if (seg == 2) {
    float acc = 0.f;
    #pragma unroll
    for (int r = 0; r < RANK; ++r) acc += wvB[n * RANK + r] * wvA[r * DIMC + k];
    wve[e] = __float2bfloat16(w_qkv[2 * (1 << 20) + e] + wv_base[e] + LORA_SC * acc);
  } else {
    wob[e] = __float2bfloat16(w_out[e]);
  }
}

// ---------------------------------------------------------------------------
// Tiled WMMA GEMM:  Out[M x 1024] = (X[M x 1024] @ W^T + bias) * outScale
// Block tile 128x64, BK=64, 256 threads = 8 waves (4x2), each wave 32x32.
// LDS rows padded to 72 bf16 (36 dwords, 16B-aligned, conflict-free gather).
// Tiles staged via async global->LDS (ASYNCcnt).
// ---------------------------------------------------------------------------
template <bool OUT_BF16>
__global__ __launch_bounds__(256) void gemm_kernel(
    const __hip_bfloat16* __restrict__ X,
    const __hip_bfloat16* __restrict__ W,
    const float* __restrict__ bias,
    float outScale,
    void* __restrict__ Out) {
  __shared__ unsigned int lds[128 * 36 + 64 * 36];   // 27648 B
  unsigned int* ldsX = lds;
  unsigned int* ldsW = lds + 128 * 36;
  uint4* ldsX4 = (uint4*)ldsX;          // row stride 9 uint4
  uint4* ldsW4 = (uint4*)ldsW;

  const int tid  = threadIdx.x;
  const int wave = tid >> 5;
  const int lane = tid & 31;
  const int half = (lane >= 16) ? 1 : 0;
  const int l15  = lane & 15;
  const int wm   = wave >> 1;          // 0..3
  const int wn   = wave & 1;           // 0..1
  const int rowBase = blockIdx.y * 128;
  const int colBase = blockIdx.x * 64;

  v8f acc[2][2] = {};

  for (int k0 = 0; k0 < DIMC; k0 += 64) {
    __syncthreads();
    // Stage X tile: 128x64 bf16 = 1024 x 16B; 4 async copies per thread.
    #pragma unroll
    for (int i = 0; i < 4; ++i) {
      int v = tid + i * 256;
      int r = v >> 3, c = v & 7;
      async_ld_b128(ldsX4 + r * 9 + c,
                    X + (size_t)(rowBase + r) * DIMC + k0 + c * 8);
    }
    // Stage W tile: 64x64 bf16 = 512 x 16B; 2 async copies per thread.
    #pragma unroll
    for (int i = 0; i < 2; ++i) {
      int v = tid + i * 256;
      int r = v >> 3, c = v & 7;
      async_ld_b128(ldsW4 + r * 9 + c,
                    W + (size_t)(colBase + r) * DIMC + k0 + c * 8);
    }
    // Prefetch next k-tile into cache while this one is consumed.
    if (k0 + 64 < DIMC) {
      __builtin_prefetch(X + (size_t)(rowBase + (tid >> 1)) * DIMC +
                             (k0 + 64) + (tid & 1) * 32, 0, 1);
      if (tid < 128)
        __builtin_prefetch(W + (size_t)(colBase + (tid >> 1)) * DIMC +
                               (k0 + 64) + (tid & 1) * 32, 0, 1);
    }
    wait_async0();
    __syncthreads();

    FragA a[2][2], bw[2][2];
    #pragma unroll
    for (int sm = 0; sm < 2; ++sm) {
      int row = wm * 32 + sm * 16 + l15;
      #pragma unroll
      for (int f = 0; f < 2; ++f)
        #pragma unroll
        for (int j = 0; j < 8; ++j) {
          int kd = j + half * 8;     // contiguous K pattern (k = 16*half + s)
          a[sm][f].u[j] = ldsX[row * 36 + f * 16 + kd];
        }
    }
    #pragma unroll
    for (int sn = 0; sn < 2; ++sn) {
      int wr = wn * 32 + sn * 16 + l15;
      #pragma unroll
      for (int f = 0; f < 2; ++f)
        #pragma unroll
        for (int v = 0; v < 8; ++v)
          bw[sn][f].u[v] = ldsW[wr * 36 + f * 16 + v + half * 8];
    }
    #pragma unroll
    for (int sm = 0; sm < 2; ++sm)
      #pragma unroll
      for (int sn = 0; sn < 2; ++sn) {
        acc[sm][sn] = wmma_bf16(a[sm][0].v, bw[sn][0].v, acc[sm][sn]);
        acc[sm][sn] = wmma_bf16(a[sm][1].v, bw[sn][1].v, acc[sm][sn]);
      }
  }

  // Epilogue: C/D layout -> row = r + half*8, col = lane&15.
  #pragma unroll
  for (int sm = 0; sm < 2; ++sm) {
    #pragma unroll
    for (int sn = 0; sn < 2; ++sn) {
      int col = colBase + wn * 32 + sn * 16 + l15;
      float bv = bias ? bias[col] : 0.f;
      int rbase = rowBase + wm * 32 + sm * 16 + half * 8;
      #pragma unroll
      for (int r = 0; r < 8; ++r) {
        float val = (acc[sm][sn][r] + bv) * outScale;
        size_t o = (size_t)(rbase + r) * DIMC + col;
        if (OUT_BF16)
          ((__hip_bfloat16*)Out)[o] = __float2bfloat16(val);
        else
          ((float*)Out)[o] = val;
      }
    }
  }
}

// ---------------------------------------------------------------------------
// Flash attention per (b, h): 128 query rows per block, 64-key chunks.
// 256 threads = 8 waves, each wave owns 16 query rows. ATT_SCALE is already
// folded into Q by the Q-projection GEMM.
// LDS carve (bytes):
//   Q  128x72 bf16  @     0  (18432)
//   K   64x72 bf16  @ 18432  ( 9216)
//   Vt  64x72 bf16  @ 27648  ( 9216)   (d-major, keys padded 64->72)
//   P   8 x 16x72   @ 36864  (18432)   per-wave scratch
// ---------------------------------------------------------------------------
__global__ __launch_bounds__(256) void attn_kernel(
    const __hip_bfloat16* __restrict__ Qb,
    const __hip_bfloat16* __restrict__ Kb,
    const __hip_bfloat16* __restrict__ Vb,
    const int* __restrict__ mask,
    __hip_bfloat16* __restrict__ Ob) {
  __shared__ __align__(16) unsigned char smem[55296];
  unsigned int*   ldsQdw  = (unsigned int*)smem;                 // stride 36 dw
  uint4*          ldsQ4   = (uint4*)smem;                        // stride 9
  unsigned int*   ldsKdw  = (unsigned int*)(smem + 18432);       // stride 36 dw
  uint4*          ldsK4   = (uint4*)(smem + 18432);              // stride 9
  unsigned short* ldsVt16 = (unsigned short*)(smem + 27648);     // stride 72
  unsigned int*   ldsVtdw = (unsigned int*)(smem + 27648);       // stride 36 dw
  unsigned short* ldsP16  = (unsigned short*)(smem + 36864);     // stride 72
  unsigned int*   ldsPdw  = (unsigned int*)(smem + 36864);       // stride 36 dw

  const int tid  = threadIdx.x;
  const int wave = tid >> 5;
  const int lane = tid & 31;
  const int half = (lane >= 16) ? 1 : 0;
  const int l15  = lane & 15;
  const int b    = blockIdx.z;
  const int h    = blockIdx.y;
  const int q0   = blockIdx.x * 128;
  const size_t headOff = (size_t)h * HD;

  // Stage Q tile (128x64 bf16, already scaled by ATT_SCALE) via async copies.
  #pragma unroll
  for (int i = 0; i < 4; ++i) {
    int v = tid + i * 256;
    int r = v >> 3, c = v & 7;
    async_ld_b128(ldsQ4 + r * 9 + c,
                  Qb + (size_t)(b * SEQ + q0 + r) * DIMC + headOff + c * 8);
  }
  wait_async0();
  __syncthreads();

  // Q A-fragments (2 over D=64) in registers for the whole key loop.
  FragA qa[2];
  const int qrow = wave * 16 + l15;
  #pragma unroll
  for (int f = 0; f < 2; ++f)
    #pragma unroll
    for (int j = 0; j < 8; ++j) {
      int kd = j + half * 8;
      qa[f].u[j] = ldsQdw[qrow * 36 + f * 16 + kd];
    }

  v8f o[4] = {};
  float m_run[8], l_run[8];
  #pragma unroll
  for (int r = 0; r < 8; ++r) { m_run[r] = NEG_BIG; l_run[r] = 0.f; }

  const int* mrow = mask + b * SEQ;
  const int pbase16 = wave * 1152;   // per-wave P region (u16 units)
  const int pbase32 = wave * 576;    // same, dword units

  for (int kc = 0; kc < SEQ; kc += 64) {
    __syncthreads();
    // Stage K chunk (64x64 key-major, async) and V chunk transposed (manual).
    #pragma unroll
    for (int i = 0; i < 2; ++i) {
      int v = tid + i * 256;
      int r = v >> 3, c = v & 7;
      size_t src = (size_t)(b * SEQ + kc + r) * DIMC + headOff + c * 8;
      async_ld_b128(ldsK4 + r * 9 + c, Kb + src);
      uint4 vv = *(const uint4*)(Vb + src);
      const unsigned short* vs = (const unsigned short*)&vv;
      #pragma unroll
      for (int e = 0; e < 8; ++e)
        ldsVt16[(c * 8 + e) * 72 + r] = vs[e];
    }
    // Prefetch next chunk (waves 0-3 -> K, waves 4-7 -> V; uniform per wave).
    if (kc + 64 < SEQ) {
      const __hip_bfloat16* base = (tid < 128) ? Kb : Vb;
      int t = tid & 127;
      __builtin_prefetch(base + (size_t)(b * SEQ + kc + 64 + (t >> 1)) * DIMC +
                             headOff + (t & 1) * 32, 0, 1);
    }
    wait_async0();
    __syncthreads();

    // Scores S[16q x 64keys]: 4 key subtiles x 2 d-fragments = 8 WMMA.
    v8f s[4] = {};
    #pragma unroll
    for (int skt = 0; skt < 4; ++skt) {
      FragA kb0, kb1;
      int key = skt * 16 + l15;
      #pragma unroll
      for (int v = 0; v < 8; ++v) {
        kb0.u[v] = ldsKdw[key * 36 + v + half * 8];
        kb1.u[v] = ldsKdw[key * 36 + 16 + v + half * 8];
      }
      s[skt] = wmma_bf16(qa[0].v, kb0.v, s[skt]);
      s[skt] = wmma_bf16(qa[1].v, kb1.v, s[skt]);
    }

    // Key-padding mask (per-lane key column; scale already in Q).
    float sv[4][8];
    #pragma unroll
    for (int skt = 0; skt < 4; ++skt) {
      bool valid = mrow[kc + skt * 16 + l15] != 0;
      #pragma unroll
      for (int r = 0; r < 8; ++r)
        sv[skt][r] = valid ? s[skt][r] : NEG_BIG;
    }

    // Online softmax per row (rows r / r+8 live in lane halves).
    #pragma unroll
    for (int r = 0; r < 8; ++r) {
      float loc = rmax16(fmaxf(fmaxf(sv[0][r], sv[1][r]),
                               fmaxf(sv[2][r], sv[3][r])));
      float mn  = fmaxf(m_run[r], loc);
      float scl = __expf(m_run[r] - mn);
      m_run[r]  = mn;
      float p0 = __expf(sv[0][r] - mn);
      float p1 = __expf(sv[1][r] - mn);
      float p2 = __expf(sv[2][r] - mn);
      float p3 = __expf(sv[3][r] - mn);
      int prow = pbase16 + (r + half * 8) * 72 + l15;
      ldsP16[prow]      = f2bf_raw(p0);
      ldsP16[prow + 16] = f2bf_raw(p1);
      ldsP16[prow + 32] = f2bf_raw(p2);
      ldsP16[prow + 48] = f2bf_raw(p3);
      l_run[r] = l_run[r] * scl + rsum16(p0 + p1 + p2 + p3);
      #pragma unroll
      for (int dt = 0; dt < 4; ++dt) o[dt][r] *= scl;
    }
    __syncthreads();   // order P writes vs reads (uniform control flow)

    // O += P(16x64) x V(64x64): 2 P-fragments x 4 d-tiles = 8 WMMA.
    FragA pa[2];
    #pragma unroll
    for (int f = 0; f < 2; ++f)
      #pragma unroll
      for (int j = 0; j < 8; ++j) {
        int kd = j + half * 8;
        pa[f].u[j] = ldsPdw[pbase32 + l15 * 36 + f * 16 + kd];
      }
    #pragma unroll
    for (int dt = 0; dt < 4; ++dt) {
      #pragma unroll
      for (int f = 0; f < 2; ++f) {
        FragA vb;
        int d = dt * 16 + l15;
        #pragma unroll
        for (int v = 0; v < 8; ++v)
          vb.u[v] = ldsVtdw[d * 36 + f * 16 + v + half * 8];
        o[dt] = wmma_bf16(pa[f].v, vb.v, o[dt]);
      }
    }
  }

  // Normalize and write O as bf16 in (B, N, H*D).
  #pragma unroll
  for (int r = 0; r < 8; ++r) {
    float inv = 1.0f / l_run[r];
    int row = q0 + wave * 16 + r + half * 8;
    #pragma unroll
    for (int dt = 0; dt < 4; ++dt)
      Ob[(size_t)(b * SEQ + row) * DIMC + headOff + dt * 16 + l15] =
          __float2bfloat16(o[dt][r] * inv);
  }
}

// ---------------------------------------------------------------------------
// Launch
// ---------------------------------------------------------------------------
extern "C" void kernel_launch(void* const* d_in, const int* in_sizes, int n_in,
                              void* d_out, int out_size, void* d_ws,
                              size_t ws_size, hipStream_t stream) {
  const float* x       = (const float*)d_in[0];
  const int*   mask    = (const int*)d_in[1];
  const float* w_qkv   = (const float*)d_in[2];
  const float* wq_base = (const float*)d_in[3];
  const float* bq_base = (const float*)d_in[4];
  const float* wq_A    = (const float*)d_in[5];
  const float* wq_B    = (const float*)d_in[6];
  const float* wv_base = (const float*)d_in[7];
  const float* bv_base = (const float*)d_in[8];
  const float* wv_A    = (const float*)d_in[9];
  const float* wv_B    = (const float*)d_in[10];
  const float* w_out   = (const float*)d_in[11];
  const float* b_out   = (const float*)d_in[12];

  char* ws = (char*)d_ws;
  const size_t MB = 1024 * 1024;
  __hip_bfloat16* xb  = (__hip_bfloat16*)(ws + 0);        // 8 MB
  __hip_bfloat16* wqe = (__hip_bfloat16*)(ws + 8 * MB);   // 2 MB
  __hip_bfloat16* wke = (__hip_bfloat16*)(ws + 10 * MB);  // 2 MB
  __hip_bfloat16* wve = (__hip_bfloat16*)(ws + 12 * MB);  // 2 MB
  __hip_bfloat16* wob = (__hip_bfloat16*)(ws + 14 * MB);  // 2 MB
  __hip_bfloat16* Qb  = (__hip_bfloat16*)(ws + 16 * MB);  // 8 MB
  __hip_bfloat16* Kb  = (__hip_bfloat16*)(ws + 24 * MB);  // 8 MB
  __hip_bfloat16* Vb  = (__hip_bfloat16*)(ws + 32 * MB);  // 8 MB
  __hip_bfloat16* Ob  = (__hip_bfloat16*)(ws + 40 * MB);  // 8 MB

  // 1) x -> bf16
  cvt_bf16_kernel<<<NTOK * DIMC / 256, 256, 0, stream>>>(x, xb, NTOK * DIMC);

  // 2) effective weights (LoRA folded in), all bf16
  build_weights_kernel<<<4 * DIMC * DIMC / 256, 256, 0, stream>>>(
      w_qkv, wq_base, wq_A, wq_B, wv_base, wv_A, wv_B, w_out,
      wqe, wke, wve, wob);

  // 3) Q/K/V projections (bf16 out; ATT_SCALE folded into Q)
  dim3 ggrid(DIMC / 64, NTOK / 128);
  gemm_kernel<true><<<ggrid, 256, 0, stream>>>(xb, wqe, bq_base, ATT_SCALE,
                                               (void*)Qb);
  gemm_kernel<true><<<ggrid, 256, 0, stream>>>(xb, wke, (const float*)nullptr,
                                               1.0f, (void*)Kb);
  gemm_kernel<true><<<ggrid, 256, 0, stream>>>(xb, wve, bv_base, 1.0f,
                                               (void*)Vb);

  // 4) masked flash attention
  attn_kernel<<<dim3(SEQ / 128, HEADS, BATCH), 256, 0, stream>>>(
      Qb, Kb, Vb, mask, Ob);

  // 5) output projection (fp32 out + bias)
  gemm_kernel<false><<<ggrid, 256, 0, stream>>>(Ob, wob, b_out, 1.0f, d_out);
}
